// GLCA_MixAttention_88819923681998
// MI455X (gfx1250) — compile-verified, matrix operand order checked by reading, not run on hardware
//
#include <hip/hip_runtime.h>
#include <hip/hip_bf16.h>
#include <math.h>
#include <stdint.h>

typedef __attribute__((ext_vector_type(16))) _Float16 v16h;
typedef __attribute__((ext_vector_type(8)))  float    v8f;
typedef __attribute__((ext_vector_type(4)))  float    v4f;

#define B_   2
#define N_   16384
#define C_   128
#define NH_  2
#define HD_  64
#define Hh_  128
#define Ww_  128
#define Hr_  32
#define Wr_  32
#define NR_  1024          // Hr*Wr
#define SCALE_ 0.125f      // hd^-0.5 = 64^-0.5
#define LN_EPS_ 1e-05f

// ---------------------------------------------------------------------------
// WMMA fragment helpers (layouts per CDNA5 ISA §7.12.2, wave32)
// All "contiguous along K" loads use float4 (b128) vector loads.
// ---------------------------------------------------------------------------

// A (16x32 f16) from row-major f32, row stride lda (lda multiple of 4)
__device__ __forceinline__ v16h load_a_f32(const float* A, int lda, int k0) {
  int l = threadIdx.x & 31; int m = l & 15; int hf = l >> 4;
  const float* row = A + (size_t)m * lda + k0 + hf * 8;
  v4f x0 = *(const v4f*)(row);
  v4f x1 = *(const v4f*)(row + 4);
  v4f x2 = *(const v4f*)(row + 16);
  v4f x3 = *(const v4f*)(row + 20);
  v16h a;
#pragma unroll
  for (int j = 0; j < 4; ++j) {
    a[j]      = (_Float16)x0[j];
    a[4 + j]  = (_Float16)x1[j];
    a[8 + j]  = (_Float16)x2[j];
    a[12 + j] = (_Float16)x3[j];
  }
  return a;
}

// B (32x16 f16) from row-major W[K][N] (column-strided; scalar loads)
__device__ __forceinline__ v16h load_b_f32(const float* Bm, int ldb, int k0, int n0) {
  int l = threadIdx.x & 31; int n = l & 15; int hf = l >> 4;
  const float* col = Bm + (size_t)(k0 + hf * 8) * ldb + n0 + n;
  v16h b;
#pragma unroll
  for (int j = 0; j < 8; ++j) b[j]     = (_Float16)col[(size_t)j * ldb];
#pragma unroll
  for (int j = 0; j < 8; ++j) b[8 + j] = (_Float16)col[(size_t)(16 + j) * ldb];
  return b;
}

// B (32x16 f16) with B[k][n] = Bt[n][k]  (Bt row-major N x K, contiguous in K)
__device__ __forceinline__ v16h load_bT_f32(const float* Bt, int ldbt, int k0, int n0) {
  int l = threadIdx.x & 31; int n = l & 15; int hf = l >> 4;
  const float* row = Bt + (size_t)(n0 + n) * ldbt + k0 + hf * 8;
  v4f x0 = *(const v4f*)(row);
  v4f x1 = *(const v4f*)(row + 4);
  v4f x2 = *(const v4f*)(row + 16);
  v4f x3 = *(const v4f*)(row + 20);
  v16h b;
#pragma unroll
  for (int j = 0; j < 4; ++j) {
    b[j]      = (_Float16)x0[j];
    b[4 + j]  = (_Float16)x1[j];
    b[8 + j]  = (_Float16)x2[j];
    b[12 + j] = (_Float16)x3[j];
  }
  return b;
}

// D (16x16 f32): VGPR r -> row (r + 8*half), lane -> col
__device__ __forceinline__ void store_d(float* D, int ldd, v8f d) {
  int l = threadIdx.x & 31; int n = l & 15; int hf = l >> 4;
#pragma unroll
  for (int r = 0; r < 8; ++r) D[(size_t)(r + 8 * hf) * ldd + n] = d[r];
}

#define WMMA_F16(a, b, c) \
  __builtin_amdgcn_wmma_f32_16x16x32_f16(false, (a), false, (b), (short)0, (c), false, false)

// ---------------------------------------------------------------------------
// Kernel 1: 4x4/stride-4 conv (HWIO) + bias + LayerNorm -> xr_ln (2048 x 128)
// ---------------------------------------------------------------------------
__global__ void srconv_ln_kernel(const float* __restrict__ x,
                                 const float* __restrict__ sr_w,
                                 const float* __restrict__ sr_b,
                                 const float* __restrict__ ln_g,
                                 const float* __restrict__ ln_b,
                                 float* __restrict__ xr_ln) {
  int pix = blockIdx.x;                // 0 .. B*Hr*Wr-1
  int b   = pix / (Hr_ * Wr_);
  int rem = pix % (Hr_ * Wr_);
  int hr = rem / Wr_, wr = rem % Wr_;
  int t = threadIdx.x;                 // channel

  __shared__ float win[16 * 128];      // 4x4 window, 128 cin
  __shared__ float red[128];

  for (int i = t; i < 16 * 128; i += 128) {
    int p = i >> 7, cin = i & 127;
    int ii = p >> 2, jj = p & 3;
    int hh = hr * 4 + ii, ww2 = wr * 4 + jj;
    win[i] = x[((size_t)b * N_ + hh * Ww_ + ww2) * C_ + cin];
  }
  __syncthreads();

  float acc = sr_b[t];
#pragma unroll 1
  for (int p = 0; p < 16; ++p) {
    const float* wcol = sr_w + (size_t)p * C_ * C_ + t;  // sr_w[p][cin][t]
    const float* xin  = win + p * 128;
#pragma unroll 8
    for (int cin = 0; cin < 128; ++cin)
      acc = fmaf(xin[cin], wcol[(size_t)cin * C_], acc);
  }

  red[t] = acc; __syncthreads();
  for (int s = 64; s > 0; s >>= 1) { if (t < s) red[t] += red[t + s]; __syncthreads(); }
  float mu = red[0] * (1.0f / 128.0f);
  __syncthreads();
  float d = acc - mu;
  red[t] = d * d; __syncthreads();
  for (int s = 64; s > 0; s >>= 1) { if (t < s) red[t] += red[t + s]; __syncthreads(); }
  float var = red[0] * (1.0f / 128.0f);
  xr_ln[(size_t)pix * C_ + t] = d * rsqrtf(var + LN_EPS_) * ln_g[t] + ln_b[t];
}

// ---------------------------------------------------------------------------
// Kernel 2: generic WMMA GEMM  D[MxN] = A[MxK] @ W[KxN] (+bias)
// 128 threads = 4 waves; each wave owns one 16x16 tile (block = 16x64)
// ---------------------------------------------------------------------------
__global__ void wmma_gemm_kernel(const float* __restrict__ A,
                                 const float* __restrict__ W,
                                 const float* __restrict__ bias,
                                 float* __restrict__ D,
                                 int lda, int ldw, int ldd, int K) {
  int wid  = threadIdx.x >> 5;
  int row0 = blockIdx.x * 16;
  int col0 = (blockIdx.y * 4 + wid) * 16;
  const float* Arow = A + (size_t)row0 * lda;

  v8f c = {};
  for (int k0 = 0; k0 < K; k0 += 32) {
    v16h a = load_a_f32(Arow, lda, k0);
    v16h b = load_b_f32(W, ldw, k0, col0);
    c = WMMA_F16(a, b, c);
  }
  if (bias) {
    float bv = bias[col0 + (threadIdx.x & 15)];
#pragma unroll
    for (int r = 0; r < 8; ++r) c[r] += bv;
  }
  store_d(D + (size_t)row0 * ldd + col0, ldd, c);
}

// ---------------------------------------------------------------------------
// Kernel 3: flash-style global attention.
// Block = 4 waves sharing one (b,h); each wave owns a 16-query tile.
// Per 32-key chunk, K (32x64 f32) and V (32x64 f32) are staged into LDS once
// per block via async global->LDS b128 copies (ASYNCcnt / s_wait_asynccnt),
// then WMMA B-fragments are fed from LDS. Online softmax with width-16
// xor-shuffle row reductions; P goes D-layout->LDS->A-layout f16.
// ---------------------------------------------------------------------------
__global__ void flash_attn_kernel(const float* __restrict__ q,
                                  const float* __restrict__ kv,  // 2048 x 256: [k | v]
                                  float* __restrict__ xg) {
  int wid = threadIdx.x >> 5;
  int qt  = blockIdx.x * 4 + wid;      // 0..1023 query tile
  int bh  = blockIdx.y;                // 0..3
  int b = bh >> 1, h = bh & 1;

  const float* Q  = q  + ((size_t)b * N_ + qt * 16) * C_ + h * HD_;
  const float* Kp = kv + (size_t)b * NR_ * 256 + h * HD_;
  const float* Vp = kv + (size_t)b * NR_ * 256 + 128 + h * HD_;

  __shared__ float    KVs[2][32 * 64];        // [0]=K chunk, [1]=V chunk (16 KB)
  __shared__ _Float16 Pl_all[4][16 * 32];     // per-wave P staging (4 KB)
  _Float16* Pl = Pl_all[wid];

  int l = threadIdx.x & 31; int lm = l & 15; int hf = l >> 4;

  // Q fragments (scaled), reused across all 32 key chunks
  v16h aq0, aq1;
  {
    const float* row = Q + (size_t)lm * C_ + hf * 8;
    v4f x0 = *(const v4f*)(row);
    v4f x1 = *(const v4f*)(row + 4);
    v4f x2 = *(const v4f*)(row + 16);
    v4f x3 = *(const v4f*)(row + 20);
    v4f y0 = *(const v4f*)(row + 32);
    v4f y1 = *(const v4f*)(row + 36);
    v4f y2 = *(const v4f*)(row + 48);
    v4f y3 = *(const v4f*)(row + 52);
#pragma unroll
    for (int j = 0; j < 4; ++j) {
      aq0[j]      = (_Float16)(x0[j] * SCALE_);
      aq0[4 + j]  = (_Float16)(x1[j] * SCALE_);
      aq0[8 + j]  = (_Float16)(x2[j] * SCALE_);
      aq0[12 + j] = (_Float16)(x3[j] * SCALE_);
      aq1[j]      = (_Float16)(y0[j] * SCALE_);
      aq1[4 + j]  = (_Float16)(y1[j] * SCALE_);
      aq1[8 + j]  = (_Float16)(y2[j] * SCALE_);
      aq1[12 + j] = (_Float16)(y3[j] * SCALE_);
    }
  }

  float mrun[8], lrun[8];
  v8f acc[4] = {};
#pragma unroll
  for (int r = 0; r < 8; ++r) { mrun[r] = -3.0e38f; lrun[r] = 0.0f; }

  for (int kc = 0; kc < NR_; kc += 32) {
    const float* Kc = Kp + (size_t)kc * 256;
    const float* Vc = Vp + (size_t)kc * 256;

    // ---- cooperative async stage of K/V chunk into LDS (1024 b128 copies) ----
    {
      unsigned t = threadIdx.x;
#pragma unroll
      for (int it = 0; it < 8; ++it) {
        unsigned i  = t + 128u * it;           // float4 index, 0..1023
        unsigned wv = i >> 9;                  // 0 = K, 1 = V
        unsigned r  = (i >> 4) & 31;           // row within chunk
        unsigned c4 = i & 15;                  // float4 within row
        const float* gsrc = (wv ? Vc : Kc) + (size_t)r * 256 + c4 * 4;
        unsigned lds = (unsigned)(uintptr_t)&KVs[wv][r * 64 + c4 * 4];
        asm volatile("global_load_async_to_lds_b128 %0, %1, off"
                     :: "v"(lds), "v"(gsrc) : "memory");
      }
      asm volatile("s_wait_asynccnt 0x0" ::: "memory");
      __syncthreads();
    }

    // ---- S tiles from LDS: keys [0,16) and [16,32) of chunk ----
    v8f s0 = {}, s1 = {};
    s0 = WMMA_F16(aq0, load_bT_f32(&KVs[0][0], 64, 0,  0),  s0);
    s0 = WMMA_F16(aq1, load_bT_f32(&KVs[0][0], 64, 32, 0),  s0);
    s1 = WMMA_F16(aq0, load_bT_f32(&KVs[0][0], 64, 0,  16), s1);
    s1 = WMMA_F16(aq1, load_bT_f32(&KVs[0][0], 64, 32, 16), s1);

    // ---- online softmax (row = r + 8*half; width-16 xor shuffles) ----
    v8f p0, p1;
#pragma unroll
    for (int r = 0; r < 8; ++r) {
      float v = fmaxf(s0[r], s1[r]);
#pragma unroll
      for (int off = 1; off < 16; off <<= 1) v = fmaxf(v, __shfl_xor(v, off, 16));
      float mnew = fmaxf(mrun[r], v);
      p0[r] = __expf(s0[r] - mnew);
      p1[r] = __expf(s1[r] - mnew);
      float s = p0[r] + p1[r];
#pragma unroll
      for (int off = 1; off < 16; off <<= 1) s += __shfl_xor(s, off, 16);
      float corr = __expf(mrun[r] - mnew);
      lrun[r] = lrun[r] * corr + s;
      mrun[r] = mnew;
#pragma unroll
      for (int t = 0; t < 4; ++t) acc[t][r] *= corr;
    }

    // ---- P: D-layout f32 -> LDS -> A-layout f16 ----
#pragma unroll
    for (int r = 0; r < 8; ++r) {
      int m = r + 8 * hf;
      Pl[m * 32 + lm]      = (_Float16)p0[r];
      Pl[m * 32 + 16 + lm] = (_Float16)p1[r];
    }
    __syncthreads();
    v16h ap;
    {
      const _Float16* row = Pl + lm * 32 + hf * 8;
#pragma unroll
      for (int j = 0; j < 8; ++j) { ap[j] = row[j]; ap[8 + j] = row[16 + j]; }
    }

    // ---- O += P @ V (V chunk from LDS, 32 x 64) ----
#pragma unroll
    for (int t = 0; t < 4; ++t)
      acc[t] = WMMA_F16(ap, load_b_f32(&KVs[1][0], 64, 0, 16 * t), acc[t]);

    __syncthreads();   // protect KVs / Pl before next chunk overwrites
  }

  // finalize: divide by row sums, write x_global
#pragma unroll
  for (int r = 0; r < 8; ++r) {
    float inv = 1.0f / lrun[r];
#pragma unroll
    for (int t = 0; t < 4; ++t) acc[t][r] *= inv;
  }
  float* Og = xg + ((size_t)b * N_ + qt * 16) * C_ + h * HD_;
#pragma unroll
  for (int t = 0; t < 4; ++t) store_d(Og + 16 * t, C_, acc[t]);
}

// ---------------------------------------------------------------------------
// Kernel 4: local window attention (64 queries x 4 keys x C=128 per window).
// K=4 is far below WMMA shapes -> VALU with float4 loads.
// ---------------------------------------------------------------------------
__global__ void local_attn_kernel(const float* __restrict__ q,
                                  const float* __restrict__ kv,
                                  float* __restrict__ xl) {
  int w = blockIdx.x;                  // 0..511
  int b = w >> 8; int rem = w & 255;
  int wh = rem >> 4, ww = rem & 15;
  int t = threadIdx.x;                 // 0..63 : one query each

  __shared__ float ks[4 * 128], vs[4 * 128];
  for (int i = t; i < 512; i += 64) {
    int kk = i >> 7, c = i & 127;
    int ki = kk >> 1, kj = kk & 1;
    size_t krow = (size_t)b * NR_ + (wh * 2 + ki) * Wr_ + (ww * 2 + kj);
    ks[i] = kv[krow * 256 + c];
    vs[i] = kv[krow * 256 + 128 + c];
  }
  __syncthreads();

  int i = t >> 3, j = t & 7;
  int n = (wh * 8 + i) * Ww_ + ww * 8 + j;
  const float* qrow = q + ((size_t)b * N_ + n) * C_;

  float s[4] = {0.f, 0.f, 0.f, 0.f};
  for (int c = 0; c < 128; c += 4) {
    v4f qv = *(const v4f*)(qrow + c);
#pragma unroll
    for (int kk = 0; kk < 4; ++kk) {
      v4f kvv = *(const v4f*)(ks + kk * 128 + c);
      s[kk] = fmaf(qv[0], kvv[0], fmaf(qv[1], kvv[1], fmaf(qv[2], kvv[2], fmaf(qv[3], kvv[3], s[kk]))));
    }
  }
#pragma unroll
  for (int kk = 0; kk < 4; ++kk) s[kk] *= SCALE_;

  float mx = fmaxf(fmaxf(s[0], s[1]), fmaxf(s[2], s[3]));
  float e0 = __expf(s[0] - mx), e1 = __expf(s[1] - mx);
  float e2 = __expf(s[2] - mx), e3 = __expf(s[3] - mx);
  float inv = 1.0f / (e0 + e1 + e2 + e3);

  float* orow = xl + ((size_t)b * N_ + n) * C_;
  for (int c = 0; c < 128; c += 4) {
    v4f v0 = *(const v4f*)(vs + c);
    v4f v1 = *(const v4f*)(vs + 128 + c);
    v4f v2 = *(const v4f*)(vs + 256 + c);
    v4f v3 = *(const v4f*)(vs + 384 + c);
    v4f o;
#pragma unroll
    for (int u = 0; u < 4; ++u)
      o[u] = (e0 * v0[u] + e1 * v1[u] + e2 * v2[u] + e3 * v3[u]) * inv;
    *(v4f*)(orow + c) = o;
  }
}

// ---------------------------------------------------------------------------
// Kernel 5: per-(batch,channel) means of x_global / x_local -> z (2 x 256)
// ---------------------------------------------------------------------------
__global__ void mean_kernel(const float* __restrict__ xg,
                            const float* __restrict__ xl,
                            float* __restrict__ z) {
  int blk = blockIdx.x;                // 0..511
  int b = blk >> 8; int ch = blk & 255;
  const float* src = (ch < 128) ? xg + (size_t)b * N_ * C_ + ch
                                : xl + (size_t)b * N_ * C_ + (ch - 128);
  __shared__ float red[256];
  float s = 0.0f;
  for (int r = threadIdx.x; r < N_; r += 256) s += src[(size_t)r * C_];
  red[threadIdx.x] = s; __syncthreads();
  for (int st = 128; st > 0; st >>= 1) {
    if (threadIdx.x < st) red[threadIdx.x] += red[threadIdx.x + st];
    __syncthreads();
  }
  if (threadIdx.x == 0) z[b * 256 + ch] = red[0] * (1.0f / (float)N_);
}

// ---------------------------------------------------------------------------
// Kernel 6: gate MLP: g = sigmoid(relu(z@f1+b1)@f2+b2)
// ---------------------------------------------------------------------------
__global__ void gate_kernel(const float* __restrict__ z,
                            const float* __restrict__ f1w, const float* __restrict__ f1b,
                            const float* __restrict__ f2w, const float* __restrict__ f2b,
                            float* __restrict__ g) {
  __shared__ float zl[512];
  __shared__ float hl[128];
  int t = threadIdx.x;                 // 0..255
  zl[t] = z[t]; zl[t + 256] = z[t + 256];
  __syncthreads();
  if (t < 128) {
    int b = t >> 6, j = t & 63;
    float s = f1b[j];
    for (int i = 0; i < 256; ++i) s = fmaf(zl[b * 256 + i], f1w[i * 64 + j], s);
    hl[t] = fmaxf(s, 0.0f);
  }
  __syncthreads();
  for (int idx = t; idx < 512; idx += 256) {
    int b = idx >> 8, o = idx & 255;
    float s = f2b[o];
    for (int i = 0; i < 64; ++i) s = fmaf(hl[b * 64 + i], f2w[i * 256 + o], s);
    g[idx] = 1.0f / (1.0f + __expf(-s));
  }
}

// ---------------------------------------------------------------------------
// Kernel 7: fused = gg*x_global + gl*x_local, out = fused @ proj_w + proj_b
// gating folded into the WMMA A-fragment loader. 4 waves / block.
// ---------------------------------------------------------------------------
__global__ void fuse_proj_kernel(const float* __restrict__ xg,
                                 const float* __restrict__ xl,
                                 const float* __restrict__ g,
                                 const float* __restrict__ pw,
                                 const float* __restrict__ pb,
                                 float* __restrict__ out) {
  int wid  = threadIdx.x >> 5;
  int row0 = blockIdx.x * 16;
  int col0 = (blockIdx.y * 4 + wid) * 16;
  int b = row0 >> 14;                  // 16384 rows per batch, tile never straddles

  const float* gg = g + b * 256;       // [:128] global gate, [128:] local gate
  int l = threadIdx.x & 31; int lm = l & 15; int hf = l >> 4;
  const float* rg = xg + (size_t)(row0 + lm) * C_;
  const float* rl = xl + (size_t)(row0 + lm) * C_;

  v8f c = {};
  for (int k0 = 0; k0 < 128; k0 += 32) {
    v16h a;
#pragma unroll
    for (int half = 0; half < 2; ++half) {
      int kb = k0 + 16 * half + hf * 8;
      v4f g0 = *(const v4f*)(rg + kb);
      v4f g1 = *(const v4f*)(rg + kb + 4);
      v4f l0 = *(const v4f*)(rl + kb);
      v4f l1 = *(const v4f*)(rl + kb + 4);
#pragma unroll
      for (int j = 0; j < 4; ++j) {
        a[8 * half + j]     = (_Float16)(gg[kb + j]     * g0[j] + gg[128 + kb + j]     * l0[j]);
        a[8 * half + 4 + j] = (_Float16)(gg[kb + 4 + j] * g1[j] + gg[128 + kb + 4 + j] * l1[j]);
      }
    }
    v16h bw = load_b_f32(pw, C_, k0, col0);
    c = WMMA_F16(a, bw, c);
  }
  float bv = pb[col0 + lm];
#pragma unroll
  for (int r = 0; r < 8; ++r) c[r] += bv;
  store_d(out + (size_t)row0 * C_ + col0, C_, c);
}

// ---------------------------------------------------------------------------
// Host launch
// ---------------------------------------------------------------------------
extern "C" void kernel_launch(void* const* d_in, const int* in_sizes, int n_in,
                              void* d_out, int out_size, void* d_ws, size_t ws_size,
                              hipStream_t stream) {
  const float* x      = (const float*)d_in[0];
  const float* q_w    = (const float*)d_in[1];
  const float* kv_w   = (const float*)d_in[2];
  const float* sr_w   = (const float*)d_in[3];
  const float* sr_b   = (const float*)d_in[4];
  const float* ln_g   = (const float*)d_in[5];
  const float* ln_b   = (const float*)d_in[6];
  const float* proj_w = (const float*)d_in[7];
  const float* proj_b = (const float*)d_in[8];
  const float* f1_w   = (const float*)d_in[9];
  const float* f1_b   = (const float*)d_in[10];
  const float* f2_w   = (const float*)d_in[11];
  const float* f2_b   = (const float*)d_in[12];
  float* out = (float*)d_out;

  // workspace carve-up (floats)
  float* ws   = (float*)d_ws;
  float* qb   = ws;                    // 32768*128 = 4,194,304
  float* xrln = qb   + 4194304;        // 2048*128  =   262,144
  float* kvb  = xrln + 262144;         // 2048*256  =   524,288
  float* xg   = kvb  + 524288;         // 4,194,304
  float* xl   = xg   + 4194304;        // 4,194,304
  float* zb   = xl   + 4194304;        // 512
  float* gb   = zb   + 512;            // 512

  // 1) sr conv + LN
  srconv_ln_kernel<<<B_ * Hr_ * Wr_, 128, 0, stream>>>(x, sr_w, sr_b, ln_g, ln_b, xrln);

  // 2) q = x @ q_w        M=32768 N=128 K=128
  wmma_gemm_kernel<<<dim3(32768 / 16, 128 / 64), 128, 0, stream>>>(
      x, q_w, nullptr, qb, C_, C_, C_, C_);

  // 3) kv = xr_ln @ kv_w  M=2048 N=256 K=128
  wmma_gemm_kernel<<<dim3(2048 / 16, 256 / 64), 128, 0, stream>>>(
      xrln, kv_w, nullptr, kvb, C_, 256, 256, C_);

  // 4) global flash attention -> x_global
  flash_attn_kernel<<<dim3(N_ / 16 / 4, B_ * NH_), 128, 0, stream>>>(qb, kvb, xg);

  // 5) local window attention -> x_local
  local_attn_kernel<<<B_ * 16 * 16, 64, 0, stream>>>(qb, kvb, xl);

  // 6) channel means -> z (2 x 256)
  mean_kernel<<<B_ * 256, 256, 0, stream>>>(xg, xl, zb);

  // 7) gate MLP -> g (2 x 256)
  gate_kernel<<<1, 256, 0, stream>>>(zb, f1_w, f1_b, f2_w, f2_b, gb);

  // 8) gated fuse + output projection -> out
  fuse_proj_kernel<<<dim3(32768 / 16, 128 / 64), 128, 0, stream>>>(
      xg, xl, gb, proj_w, proj_b, out);
}
